// UBlock_60258391163528
// MI455X (gfx1250) — matrix operand order repaired and checked by reading, not compiled, not run on hardware
//
#include <hip/hip_runtime.h>

// =====================================================================
// Sparse 3D U-Net (UBlock) for MI455X (gfx1250, wave32, CDNA5 WMMA).
//
// - Every conv (3x3x3 submanifold, 1x1x1 identity, k2s2 down, k2s2
//   transpose-up) is an implicit GEMM on V_WMMA_F32_16X16X32_F16
//   (f16 in, f32 accumulate). Activations channels-last ([vox][C], f16)
//   so per-lane B fragments are 32B-contiguous; weights repacked to
//   [tap][cout][cin] f16.
// - Conv kernels fully templated (CIN/COUT/DIM/KS/PAD compile-time),
//   blockIdx.{z,y} carry the spatial row -> no runtime div/mod.
// - (z,y) is block-uniform, so per-tap COUTxCIN weight blocks are
//   DMA'd into LDS with the Tensor Data Mover, DOUBLE-BUFFERED:
//   wave0 issues tap i+1 while the block computes tap i, then waits
//   TENSORcnt<=1 (TDM completes in order). A-fragments read from LDS.
// - BN stats / masks / residuals stay f32. Workspace ~300 MB.
//
// Input order assumed (setup_inputs insertion order): x, mask, then per
// level: blocks[0..1] (g1,b1,w1,g2,b2,w2[,wi]), cg,cb,cw,dg,db,dw,
// blocks_tail[0..1]. mask is bool -> unsigned char.
// =====================================================================

typedef __attribute__((ext_vector_type(16))) _Float16 v16h;
typedef __attribute__((ext_vector_type(8)))  _Float16 v8h;
typedef __attribute__((ext_vector_type(8)))  float    v8f;
typedef __attribute__((ext_vector_type(4)))  unsigned int v4u;
typedef __attribute__((ext_vector_type(8)))  int v8i;
typedef __attribute__((ext_vector_type(4)))  int v4i;

#define TPB 256
#define BN_EPS 1e-4f

static __device__ __forceinline__ v8f wmma_f16(v16h a, v16h b, v8f c) {
  return __builtin_amdgcn_wmma_f32_16x16x32_f16(false, a, false, b,
                                                (short)0, c, false, false);
}

// A-fragment: lane holds row M = lane&15. Contiguous K layout (matches B):
// lanes 0-15 hold K in {0..15}; lanes 16-31 hold K in {16..31}.
static __device__ __forceinline__ v16h load_a(const _Float16* wrow, unsigned hi) {
  v8h lo = *(const v8h*)(wrow + hi * 16);
  v8h hh = *(const v8h*)(wrow + hi * 16 + 8);
  v16h a;
#pragma unroll
  for (int e = 0; e < 8; ++e) { a[e] = lo[e]; a[e + 8] = hh[e]; }
  return a;
}

// ---- Tensor Data Mover: 1-row tile of n f16 elements, global -> LDS ----
// D# per CDNA5 ISA ch.8: group0 = {count/type, lds_addr, global_addr},
// group1 = {data_size, tensor/tile dims, strides}; groups 2/3 zero (<=2D).
// This toolchain's builtin: 6 args (g0, g1, g2, g3, extra, cpol).
#if __has_builtin(__builtin_amdgcn_tensor_load_to_lds)
#define HAS_TDM 1
static __device__ __forceinline__ void tdm_load_row(unsigned lds_off,
                                                    const void* gsrc, int n) {
  unsigned long ga = (unsigned long)gsrc;
  v4u g0;
  g0[0] = 1u;                                   // count=1 (valid user D#)
  g0[1] = lds_off;                              // lds_addr (bytes)
  g0[2] = (unsigned)ga;                         // global_addr[31:0]
  g0[3] = (unsigned)((ga >> 32) & 0x01FFFFFFu)  // global_addr[56:32]
          | 0x80000000u;                        // type=2 ("image")
  v8i g1;
  g1[0] = 0x00010000;                           // data_size=1 (2B), no flags
  g1[1] = (n & 0xFFFF) << 16;                   // tensor_dim0[15:0]
  g1[2] = ((n >> 16) & 0xFFFF) | (1 << 16);     // tensor_dim0[31:16] | tensor_dim1=1
  g1[3] = (n & 0xFFFF) << 16;                   // tile_dim0 = n
  g1[4] = 1;                                    // tile_dim1 = 1, tile_dim2 = 0
  g1[5] = n;                                    // tensor_dim0_stride[31:0]
  g1[6] = 0;
  g1[7] = 0;
  v4i z4 = {0, 0, 0, 0};
  v8i z8 = {0, 0, 0, 0, 0, 0, 0, 0};
  __builtin_amdgcn_tensor_load_to_lds(g0, g1, z4, z4, z8, 0);
}
#else
#define HAS_TDM 0
#endif

// issue a stage of n f16 into an LDS slot (n multiple of 1024)
static __device__ __forceinline__ void stage_issue(_Float16* dst,
                                                   const _Float16* src, int n) {
#if HAS_TDM
  if (threadIdx.x < 32)                         // one wave drives the TDM
    tdm_load_row((unsigned)(unsigned long)dst, src, n);
#else
  for (int i = threadIdx.x * 8; i < n; i += 128 * 8)
    *(v8h*)(dst + i) = *(const v8h*)(src + i);
#endif
}
static __device__ __forceinline__ void stage_wait_keep1() {
#if HAS_TDM
  if (threadIdx.x < 32) __builtin_amdgcn_s_wait_tensorcnt(1);
#endif
}
static __device__ __forceinline__ void stage_wait_all() {
#if HAS_TDM
  if (threadIdx.x < 32) __builtin_amdgcn_s_wait_tensorcnt(0);
#endif
}

// ------------------------- elementwise kernels -------------------------

__global__ void k_mask_f32(const unsigned char* __restrict__ m, float* __restrict__ o, int n) {
  int i = blockIdx.x * blockDim.x + threadIdx.x;
  if (i < n) o[i] = m[i] ? 1.f : 0.f;
}

__global__ void k_down_mask(const float* __restrict__ mf, float* __restrict__ mc,
                            int Dc, int Hc, int Wc) {
  int n = Dc * Hc * Wc;
  int i = blockIdx.x * blockDim.x + threadIdx.x;
  if (i >= n) return;
  int xc = i % Wc; int t = i / Wc; int yc = t % Hc; int zc = t / Hc;
  int Hf = Hc * 2, Wf = Wc * 2;
  float mx = 0.f;
  for (int dz = 0; dz < 2; ++dz)
    for (int dy = 0; dy < 2; ++dy)
      for (int dx = 0; dx < 2; ++dx)
        mx = fmaxf(mx, mf[((2 * zc + dz) * Hf + 2 * yc + dy) * Wf + 2 * xc + dx]);
  mc[i] = mx;
}

__global__ void k_count(const float* __restrict__ m, int n, float* __restrict__ out) {
  __shared__ float sh[TPB];
  float s = 0.f;
  for (int i = threadIdx.x; i < n; i += blockDim.x) s += m[i];
  sh[threadIdx.x] = s; __syncthreads();
  for (int o = TPB / 2; o > 0; o >>= 1) {
    if ((int)threadIdx.x < o) sh[threadIdx.x] += sh[threadIdx.x + o];
    __syncthreads();
  }
  if (threadIdx.x == 0) out[0] = sh[0];
}

__global__ void k_from_nchw_mask(const float* __restrict__ x, const float* __restrict__ m,
                                 float* __restrict__ o, int C, int Nv) {
  int i = blockIdx.x * blockDim.x + threadIdx.x;
  if (i >= Nv * C) return;
  int c = i % C, v = i / C;
  o[i] = x[c * Nv + v] * m[v];
}

__global__ void k_to_nchw(const float* __restrict__ x, float* __restrict__ o, int C, int Nv) {
  int i = blockIdx.x * blockDim.x + threadIdx.x;
  if (i >= Nv * C) return;
  int c = i % C, v = i / C;
  o[c * Nv + v] = x[i];
}

__global__ void k_concat_copy(const float* __restrict__ x, float* __restrict__ cat,
                              int C, int Nv) {
  int i = blockIdx.x * blockDim.x + threadIdx.x;
  if (i >= Nv * C) return;
  int c = i % C, v = i / C;
  cat[v * 2 * C + c] = x[i];
}

__global__ void k_cast_f16(const float* __restrict__ x, _Float16* __restrict__ o, int n) {
  int i = blockIdx.x * blockDim.x + threadIdx.x;
  if (i < n) o[i] = (_Float16)x[i];
}

__global__ void k_bn_stats(const float* __restrict__ x, int C, int Nv,
                           float* __restrict__ sums, float* __restrict__ sumsq) {
  __shared__ float sh[TPB], shq[TPB];
  int c = blockIdx.x;
  float s = 0.f, q = 0.f;
  for (int v = threadIdx.x; v < Nv; v += blockDim.x) {
    float t = x[v * C + c];   // x is exactly zero at inactive voxels
    s += t; q += t * t;
  }
  sh[threadIdx.x] = s; shq[threadIdx.x] = q; __syncthreads();
  for (int o = TPB / 2; o > 0; o >>= 1) {
    if ((int)threadIdx.x < o) { sh[threadIdx.x] += sh[threadIdx.x + o]; shq[threadIdx.x] += shq[threadIdx.x + o]; }
    __syncthreads();
  }
  if (threadIdx.x == 0) { sums[c] = sh[0]; sumsq[c] = shq[0]; }
}

__global__ void k_bn_relu_f16(const float* __restrict__ x, const float* __restrict__ m,
                              const float* __restrict__ g, const float* __restrict__ b,
                              const float* __restrict__ sums, const float* __restrict__ sumsq,
                              const float* __restrict__ cnt, int C, int tot,
                              _Float16* __restrict__ o) {
  int i = blockIdx.x * blockDim.x + threadIdx.x;
  if (i >= tot) return;
  int c = i % C, v = i / C;
  float ct = cnt[0];
  float mu = sums[c] / ct;
  float var = sumsq[c] / ct - mu * mu;
  float inv = rsqrtf(var + BN_EPS);
  float y = (x[i] - mu) * inv * g[c] + b[c];
  o[i] = (_Float16)(fmaxf(y, 0.f) * m[v]);
}

// repack f32 weights -> f16 [tap][cout][cin]; io=0: src OIDHW, io=1: src IODHW
__global__ void k_repack_w(const float* __restrict__ w, _Float16* __restrict__ wh,
                           int Cout, int Cin, int kvol, int io) {
  int tot = Cout * Cin * kvol;
  int i = blockIdx.x * blockDim.x + threadIdx.x;
  if (i >= tot) return;
  int ci = i % Cin; int t = i / Cin;
  int co = t % Cout; int tap = t / Cout;
  int s = io ? ((ci * Cout + co) * kvol + tap)
             : ((co * Cin + ci) * kvol + tap);
  wh[i] = (_Float16)w[s];
}

// ------------------------- WMMA conv kernels -------------------------
// act: [DIM][DIM][DIM][CIN] f16  wt: [tap][COUT][CIN] f16
// out: [vox][COUT] f32 = conv * mask (+ res). grid = (2, DIM, DIM).
// Double-buffered TDM weight pipeline over the (contiguous) valid taps.
template <int CIN, int COUT, int DIM, int KS, int PAD>
__global__ __launch_bounds__(128) void k_conv_wmma(
    const _Float16* __restrict__ act, const _Float16* __restrict__ wt,
    const float* __restrict__ mask, const float* __restrict__ res,
    float* __restrict__ out) {
  constexpr int nCo = COUT / 16, nX = DIM / 16, WBLK = COUT * CIN;
  __shared__ _Float16 wlds[2][WBLK];              // double-buffered tap weights
  const unsigned wave = threadIdx.x >> 5, lane = threadIdx.x & 31;
  const unsigned hi = lane >> 4, nn = lane & 15;
  unsigned tile = blockIdx.x * 4 + wave;
  const bool active = tile < (unsigned)(nCo * nX);
  if (!active) tile = 0;
  const unsigned co0 = (tile % nCo) * 16;         // nCo power of two
  const unsigned x0 = (tile / nCo) * 16;
  const int y = blockIdx.y, z = blockIdx.z;       // block-uniform
  // valid tap rectangle: kd in [d0,d1), kh in [h0,h1), kw in [0,KS)
  const int d0 = (PAD - z) > 0 ? (PAD - z) : 0;
  const int d1 = (DIM + PAD - z) < KS ? (DIM + PAD - z) : KS;
  const int h0 = (PAD - y) > 0 ? (PAD - y) : 0;
  const int h1 = (DIM + PAD - y) < KS ? (DIM + PAD - y) : KS;
  const int nt = (d1 - d0) * (h1 - h0) * KS;      // >=1 always (PAD<KS)
  int kd = d0, kh = h0, kw = 0;
  stage_issue(wlds[0], wt + ((kd * KS + kh) * KS + kw) * WBLK, WBLK);
  v8f acc = {};
  for (int i = 0; i < nt; ++i) {
    // next tap coordinates (uniform carry logic)
    int nkw = kw + 1, nkh = kh, nkd = kd;
    if (nkw == KS) { nkw = 0; if (++nkh == h1) { nkh = h0; ++nkd; } }
    if (i + 1 < nt) {                             // issue ahead into other slot
      stage_issue(wlds[(i + 1) & 1], wt + ((nkd * KS + nkh) * KS + nkw) * WBLK, WBLK);
      stage_wait_keep1();                         // slot i complete (in-order TDM)
    } else {
      stage_wait_all();
    }
    __syncthreads();                              // slot i visible to all waves
    const int zi = z + kd - PAD, yi = y + kh - PAD;
    const int xi = (int)(x0 + nn) + kw - PAD;
    const bool xv = (xi >= 0) && (xi < DIM);
    const _Float16* arow =
        act + ((zi * DIM + yi) * DIM + (xv ? xi : 0)) * CIN + hi * 16;
    const _Float16* wrow = &wlds[i & 1][(co0 + nn) * CIN];
    if (yi + 1 < DIM)
      __builtin_prefetch(act + ((zi * DIM + yi + 1) * DIM + (int)x0) * CIN, 0, 1);
#pragma unroll
    for (int cb = 0; cb < CIN / 32; ++cb) {
      v16h b = {};
      if (xv) b = *(const v16h*)(arow + cb * 32);  // 32B contiguous
      v16h a = load_a(wrow + cb * 32, hi);         // ds_load from LDS
      acc = wmma_f16(a, b, acc);
    }
    __syncthreads();                              // consumed before overwrite
    kd = nkd; kh = nkh; kw = nkw;
  }
  if (active) {
    const int vox = (z * DIM + y) * DIM + (int)(x0 + nn);
    const float m = mask[vox];
    const int ob = vox * COUT + (int)(co0 + hi * 8);
#pragma unroll
    for (int r = 0; r < 8; ++r) {
      float v = acc[r] * m;
      if (res) v += res[ob + r];
      out[ob + r] = v;
    }
  }
}

// k=2 s=2 down-conv: fine [2DIMC]^3 [CIN] f16 -> coarse [DIMC]^3 [COUT] f32
template <int CIN, int COUT, int DIMC>
__global__ __launch_bounds__(128) void k_conv_down_wmma(
    const _Float16* __restrict__ act, const _Float16* __restrict__ wt,
    const float* __restrict__ maskc, float* __restrict__ out) {
  constexpr int nCo = COUT / 16, nX = DIMC / 16, DIMF = DIMC * 2;
  const unsigned wave = threadIdx.x >> 5, lane = threadIdx.x & 31;
  const unsigned hi = lane >> 4, nn = lane & 15;
  unsigned tile = blockIdx.x * 4 + wave;
  const bool active = tile < (unsigned)(nCo * nX);
  if (!active) tile = 0;
  const unsigned co0 = (tile % nCo) * 16;
  const unsigned x0 = (tile / nCo) * 16;
  const int yc = blockIdx.y, zc = blockIdx.z;
  v8f acc = {};
#pragma unroll
  for (int kd = 0; kd < 2; ++kd)
#pragma unroll
    for (int kh = 0; kh < 2; ++kh)
#pragma unroll
      for (int kw = 0; kw < 2; ++kw) {
        const int zi = 2 * zc + kd, yi = 2 * yc + kh, xi = 2 * (int)(x0 + nn) + kw;
        const _Float16* arow = act + ((zi * DIMF + yi) * DIMF + xi) * CIN + hi * 16;
        const int tap = (kd * 2 + kh) * 2 + kw;
        const _Float16* wrow = wt + (tap * COUT + (int)(co0 + nn)) * CIN;
#pragma unroll
        for (int cb = 0; cb < CIN / 32; ++cb) {
          v16h b = *(const v16h*)(arow + cb * 32);
          v16h a = load_a(wrow + cb * 32, hi);
          acc = wmma_f16(a, b, acc);
        }
      }
  if (active) {
    const int vox = (zc * DIMC + yc) * DIMC + (int)(x0 + nn);
    const float m = maskc[vox];
    const int ob = vox * COUT + (int)(co0 + hi * 8);
#pragma unroll
    for (int r = 0; r < 8; ++r) out[ob + r] = acc[r] * m;
  }
}

// k=2 s=2 transpose-conv: coarse [DIMC]^3 [CIN] f16 -> fine voxels; one tap
// per tile (fixed output parity). Writes [vox][OSTR] at channel OOFF.
template <int CIN, int COUT, int DIMC, int OSTR, int OOFF>
__global__ __launch_bounds__(128) void k_conv_up_wmma(
    const _Float16* __restrict__ actc, const _Float16* __restrict__ wt,
    const float* __restrict__ maskf, float* __restrict__ out) {
  constexpr int nCo = COUT / 16, nXc = DIMC / 16, DIMF = DIMC * 2;
  const unsigned wave = threadIdx.x >> 5, lane = threadIdx.x & 31;
  const unsigned hi = lane >> 4, nn = lane & 15;
  unsigned tile = blockIdx.x * 4 + wave;
  const bool active = tile < (unsigned)(nCo * 2 * nXc);
  if (!active) tile = 0;
  const unsigned co0 = (tile % nCo) * 16; tile /= nCo;
  const unsigned px = tile & 1;
  const unsigned xc0 = (tile >> 1) * 16;
  const int yf = blockIdx.y, zf = blockIdx.z;
  const int zc = zf >> 1, yc = yf >> 1;
  const int tap = ((zf & 1) * 2 + (yf & 1)) * 2 + (int)px;
  const _Float16* arow = actc + ((zc * DIMC + yc) * DIMC + (int)(xc0 + nn)) * CIN + hi * 16;
  const _Float16* wrow = wt + (tap * COUT + (int)(co0 + nn)) * CIN;
  v8f acc = {};
#pragma unroll
  for (int cb = 0; cb < CIN / 32; ++cb) {
    v16h b = *(const v16h*)(arow + cb * 32);
    v16h a = load_a(wrow + cb * 32, hi);
    acc = wmma_f16(a, b, acc);
  }
  if (active) {
    const int vox = (zf * DIMF + yf) * DIMF + 2 * (int)(xc0 + nn) + (int)px;
    const float m = maskf[vox];
    const int ob = vox * OSTR + OOFF + (int)(co0 + hi * 8);
#pragma unroll
    for (int r = 0; r < 8; ++r) out[ob + r] = acc[r] * m;
  }
}

// ------------------------- host orchestration -------------------------

static inline int cdiv(long a, long b) { return (int)((a + b - 1) / b); }

struct ResW { const float *g1, *b1, *w1, *g2, *b2, *w2, *wi; };
struct Scratch { float* sums; float* sumsq; _Float16* wh; hipStream_t st; };

static void bn_relu(Scratch& S, const float* x, const float* m, const float* g,
                    const float* b, const float* cnt, int C, int Nv, _Float16* h) {
  k_bn_stats<<<C, TPB, 0, S.st>>>(x, C, Nv, S.sums, S.sumsq);
  k_bn_relu_f16<<<cdiv((long)Nv * C, TPB), TPB, 0, S.st>>>(x, m, g, b, S.sums, S.sumsq,
                                                           cnt, C, Nv * C, h);
}

template <int CIN, int COUT, int DIM, int KS, int PAD>
static void conv_t(Scratch& S, const _Float16* act, const float* w,
                   const float* mask, const float* res, float* out) {
  constexpr int KVOL = KS * KS * KS;
  k_repack_w<<<cdiv(COUT * CIN * KVOL, TPB), TPB, 0, S.st>>>(w, S.wh, COUT, CIN, KVOL, 0);
  dim3 grid(cdiv((COUT / 16) * (DIM / 16), 4), DIM, DIM);
  k_conv_wmma<CIN, COUT, DIM, KS, PAD><<<grid, 128, 0, S.st>>>(act, S.wh, mask, res, out);
}

template <int CIN, int COUT, int DIMC>
static void down_t(Scratch& S, const _Float16* act, const float* w,
                   const float* maskc, float* out) {
  k_repack_w<<<cdiv(COUT * CIN * 8, TPB), TPB, 0, S.st>>>(w, S.wh, COUT, CIN, 8, 0);
  dim3 grid(cdiv((COUT / 16) * (DIMC / 16), 4), DIMC, DIMC);
  k_conv_down_wmma<CIN, COUT, DIMC><<<grid, 128, 0, S.st>>>(act, S.wh, maskc, out);
}

template <int CIN, int COUT, int DIMC, int OSTR, int OOFF>
static void up_t(Scratch& S, const _Float16* actc, const float* w,
                 const float* maskf, float* out) {
  k_repack_w<<<cdiv(COUT * CIN * 8, TPB), TPB, 0, S.st>>>(w, S.wh, COUT, CIN, 8, 1);
  dim3 grid(cdiv((COUT / 16) * 2 * (DIMC / 16), 4), DIMC * 2, DIMC * 2);
  k_conv_up_wmma<CIN, COUT, DIMC, OSTR, OOFF><<<grid, 128, 0, S.st>>>(actc, S.wh, maskf, out);
}

template <int CIN, int COUT, int DIM, bool HASWI>
static void res_block_t(Scratch& S, const float* xin, float* xout, const float* m,
                        const float* cnt, const ResW& p, _Float16* hbuf,
                        float* idn, float* tmp) {
  constexpr int Nv = DIM * DIM * DIM;
  const float* resptr = xin;
  if constexpr (HASWI) {   // 1x1x1 submanifold conv on identity branch
    k_cast_f16<<<cdiv((long)Nv * CIN, TPB), TPB, 0, S.st>>>(xin, hbuf, Nv * CIN);
    conv_t<CIN, COUT, DIM, 1, 0>(S, hbuf, p.wi, m, nullptr, idn);
    resptr = idn;
  }
  bn_relu(S, xin, m, p.g1, p.b1, cnt, CIN, Nv, hbuf);
  conv_t<CIN, COUT, DIM, 3, 1>(S, hbuf, p.w1, m, nullptr, tmp);
  bn_relu(S, tmp, m, p.g2, p.b2, cnt, COUT, Nv, hbuf);
  conv_t<COUT, COUT, DIM, 3, 1>(S, hbuf, p.w2, m, resptr, xout);
}

extern "C" void kernel_launch(void* const* d_in, const int* in_sizes, int n_in,
                              void* d_out, int out_size, void* d_ws, size_t ws_size,
                              hipStream_t stream) {
  (void)in_sizes; (void)n_in; (void)out_size; (void)ws_size;
  const int N0 = 64 * 64 * 64, N1 = 32 * 32 * 32, N2 = 16 * 16 * 16;

  // ---- inputs ----
  const float* x_in = (const float*)d_in[0];                      // [32][64^3] f32
  const unsigned char* mask_in = (const unsigned char*)d_in[1];   // bool [64^3]
  int pi = 2;
  auto nx = [&]() { return (const float*)d_in[pi++]; };
  auto rd_res = [&](bool wi) {
    ResW r; r.g1 = nx(); r.b1 = nx(); r.w1 = nx(); r.g2 = nx(); r.b2 = nx(); r.w2 = nx();
    r.wi = wi ? nx() : nullptr; return r;
  };
  ResW L0b0 = rd_res(false), L0b1 = rd_res(false);
  const float *cg0 = nx(), *cb0 = nx(), *cw0 = nx(), *dg0 = nx(), *db0 = nx(), *dw0 = nx();
  ResW L0t0 = rd_res(true), L0t1 = rd_res(false);
  ResW L1b0 = rd_res(false), L1b1 = rd_res(false);
  const float *cg1 = nx(), *cb1 = nx(), *cw1 = nx(), *dg1 = nx(), *db1 = nx(), *dw1 = nx();
  ResW L1t0 = rd_res(true), L1t1 = rd_res(false);
  ResW L2b0 = rd_res(false), L2b1 = rd_res(false);

  // ---- workspace carve-up (~300 MB) ----
  char* base = (char*)d_ws; size_t off = 0;
  auto alloc = [&](size_t bytes) { void* p = base + off; off = (off + bytes + 255) & ~(size_t)255; return p; };
  float* m0 = (float*)alloc((size_t)N0 * 4);
  float* m1 = (float*)alloc((size_t)N1 * 4);
  float* m2 = (float*)alloc((size_t)N2 * 4);
  float* cnt = (float*)alloc(16);
  float* sums = (float*)alloc(128 * 4);
  float* sumsq = (float*)alloc(128 * 4);
  float* x0a = (float*)alloc((size_t)N0 * 32 * 4);
  float* x0b = (float*)alloc((size_t)N0 * 32 * 4);
  float* tmp0 = (float*)alloc((size_t)N0 * 32 * 4);
  float* idn0 = (float*)alloc((size_t)N0 * 32 * 4);
  float* cat0 = (float*)alloc((size_t)N0 * 64 * 4);
  _Float16* h0h = (_Float16*)alloc((size_t)N0 * 64 * 2);
  float* x1a = (float*)alloc((size_t)N1 * 64 * 4);
  float* x1b = (float*)alloc((size_t)N1 * 64 * 4);
  float* tmp1 = (float*)alloc((size_t)N1 * 64 * 4);
  float* idn1 = (float*)alloc((size_t)N1 * 64 * 4);
  float* cat1 = (float*)alloc((size_t)N1 * 128 * 4);
  _Float16* h1h = (_Float16*)alloc((size_t)N1 * 128 * 2);
  float* x2a = (float*)alloc((size_t)N2 * 128 * 4);
  float* x2b = (float*)alloc((size_t)N2 * 128 * 4);
  float* tmp2 = (float*)alloc((size_t)N2 * 128 * 4);
  _Float16* h2h = (_Float16*)alloc((size_t)N2 * 128 * 2);
  _Float16* wh = (_Float16*)alloc(442368L * 2);   // largest repacked weight

  Scratch S{sums, sumsq, wh, stream};

  // ---- masks, counts, input layout ----
  k_mask_f32<<<cdiv(N0, TPB), TPB, 0, stream>>>(mask_in, m0, N0);
  k_down_mask<<<cdiv(N1, TPB), TPB, 0, stream>>>(m0, m1, 32, 32, 32);
  k_down_mask<<<cdiv(N2, TPB), TPB, 0, stream>>>(m1, m2, 16, 16, 16);
  k_count<<<1, TPB, 0, stream>>>(m0, N0, cnt + 0);
  k_count<<<1, TPB, 0, stream>>>(m1, N1, cnt + 1);
  k_count<<<1, TPB, 0, stream>>>(m2, N2, cnt + 2);
  k_from_nchw_mask<<<cdiv((long)N0 * 32, TPB), TPB, 0, stream>>>(x_in, m0, x0a, 32, N0);

  // ---- level 0 blocks ----
  res_block_t<32, 32, 64, false>(S, x0a, x0b, m0, cnt + 0, L0b0, h0h, nullptr, tmp0);
  res_block_t<32, 32, 64, false>(S, x0b, x0a, m0, cnt + 0, L0b1, h0h, nullptr, tmp0);

  // ---- down L0 -> L1 ----
  bn_relu(S, x0a, m0, cg0, cb0, cnt + 0, 32, N0, h0h);
  down_t<32, 64, 32>(S, h0h, cw0, m1, x1a);

  // ---- level 1 blocks ----
  res_block_t<64, 64, 32, false>(S, x1a, x1b, m1, cnt + 1, L1b0, h1h, nullptr, tmp1);
  res_block_t<64, 64, 32, false>(S, x1b, x1a, m1, cnt + 1, L1b1, h1h, nullptr, tmp1);

  // ---- down L1 -> L2 ----
  bn_relu(S, x1a, m1, cg1, cb1, cnt + 1, 64, N1, h1h);
  down_t<64, 128, 16>(S, h1h, cw1, m2, x2a);

  // ---- level 2 blocks ----
  res_block_t<128, 128, 16, false>(S, x2a, x2b, m2, cnt + 2, L2b0, h2h, nullptr, tmp2);
  res_block_t<128, 128, 16, false>(S, x2b, x2a, m2, cnt + 2, L2b1, h2h, nullptr, tmp2);

  // ---- up L2 -> L1, concat, tail ----
  bn_relu(S, x2a, m2, dg1, db1, cnt + 2, 128, N2, h2h);
  up_t<128, 64, 16, 128, 64>(S, h2h, dw1, m1, cat1);
  k_concat_copy<<<cdiv((long)N1 * 64, TPB), TPB, 0, stream>>>(x1a, cat1, 64, N1);
  res_block_t<128, 64, 32, true>(S, cat1, x1b, m1, cnt + 1, L1t0, h1h, idn1, tmp1);
  res_block_t<64, 64, 32, false>(S, x1b, x1a, m1, cnt + 1, L1t1, h1h, nullptr, tmp1);

  // ---- up L1 -> L0, concat, tail ----
  bn_relu(S, x1a, m1, dg0, db0, cnt + 1, 64, N1, h1h);
  up_t<64, 32, 32, 64, 32>(S, h1h, dw0, m0, cat0);
  k_concat_copy<<<cdiv((long)N0 * 32, TPB), TPB, 0, stream>>>(x0a, cat0, 32, N0);
  res_block_t<64, 32, 64, true>(S, cat0, x0b, m0, cnt + 0, L0t0, h0h, idn0, tmp0);
  res_block_t<32, 32, 64, false>(S, x0b, x0a, m0, cnt + 0, L0t1, h0h, nullptr, tmp0);

  // ---- output: channels-last -> NCDHW ----
  k_to_nchw<<<cdiv((long)N0 * 32, TPB), TPB, 0, stream>>>(x0a, (float*)d_out, 32, N0);
}
